// Transformer_5746666242155
// MI455X (gfx1250) — compile-verified, matrix operand order checked by reading, not run on hardware
//
#include <hip/hip_runtime.h>
#include <math.h>

// ---------------- problem constants ----------------
#define B_   64
#define S_   128
#define D_   512
#define H_   8
#define DK_  64
#define L_   6
#define DFF_ 2048
#define AH_  2

typedef __attribute__((ext_vector_type(16))) _Float16 v16h;
typedef __attribute__((ext_vector_type(8)))  _Float16 v8h;
typedef __attribute__((ext_vector_type(8)))  float    v8f;

union FragH { v16h v; v8h h8[2]; _Float16 h[16]; };
union FragF { v8f  v; float f[8]; };

// ---------------------------------------------------------------------------
// Fragment loader. CDNA5 16-bit A-matrix 16x32 layout (wave32):
//   lanes 0-15 : row M = lane,    elems 0..7 -> K=0..7,  elems 8..15 -> K=16..23
//   lanes16-31 : row M = lane-16, elems 0..7 -> K=8..15, elems 8..15 -> K=24..31
// B fragment (32x16) mirrors this with lane = column N. All operands are kept
// in [rows, K] row-major layout (activations natural, weights pre-transposed,
// V transposed by the V-projection epilogue), so this is the only loader and
// every fragment load is two contiguous global_load_b128.
// ---------------------------------------------------------------------------
__device__ inline v16h load_frag_k16(const _Float16* base, int ld, int rc, int k0, int lane) {
  int r  = rc + (lane & 15);
  int kg = (lane >> 4) & 1;
  const _Float16* p = base + (size_t)r * (size_t)ld + (size_t)(k0 + kg * 8);
  FragH f;
  f.h8[0] = *(const v8h*)(p);
  f.h8[1] = *(const v8h*)(p + 16);
  return f.v;
}

// ---------------------------------------------------------------------------
// Generic batched WMMA GEMM:  C[z] = act( A[z] * B[z] (+bias) (+resid) )
// 256 threads = 8 waves; waves tiled 2(M) x 4(N); each wave -> 32x32 output
// via 2x2 v8f accumulators; K-step 32 with v_wmma_f32_16x16x32_f16.
// Batch index z decomposed z = z1*Z2 + z2 with independent strides so
// per-(batch,head) sub-matrices of [B*S, H*DK] buffers are addressable.
// storeCT: write the f16 result transposed (Ch[col*ldc + row]) — used by the
// V projection so attn@V can use the contiguous fragment loader.
// ---------------------------------------------------------------------------
__global__ __launch_bounds__(256)
void k_gemm_wmma(const _Float16* __restrict__ A, const _Float16* __restrict__ Bt,
                 const float* __restrict__ bias, const float* __restrict__ resid,
                 float* __restrict__ Cf, _Float16* __restrict__ Ch,
                 int M, int N, int K, int lda, int ldb, int ldc, int Z2,
                 long long sA1, long long sA2, long long sB1, long long sB2,
                 long long sC1, long long sC2, int storeCT, int doRelu)
{
  int z  = blockIdx.z;
  int z1 = z / Z2, z2 = z % Z2;
  const _Float16* Az = A  + z1 * sA1 + z2 * sA2;
  const _Float16* Bz = Bt + z1 * sB1 + z2 * sB2;
  long long cOff = z1 * sC1 + z2 * sC2;

  int wave = threadIdx.x >> 5;
  int lane = threadIdx.x & 31;
  int wm = wave >> 2, wn = wave & 3;
  int row0 = blockIdx.y * 64 + wm * 32;
  int col0 = blockIdx.x * 128 + wn * 32;
  if (row0 >= M || col0 >= N) return;   // wave-uniform; EXEC stays full for WMMA

  FragF acc[2][2];
#pragma unroll
  for (int i = 0; i < 2; ++i)
#pragma unroll
    for (int j = 0; j < 2; ++j)
#pragma unroll
      for (int r = 0; r < 8; ++r) acc[i][j].f[r] = 0.0f;

  int rA  = lane & 15;
  int kgo = ((lane >> 4) & 1) * 8;

  for (int k0 = 0; k0 < K; k0 += 32) {
    v16h a0 = load_frag_k16(Az, lda, row0,      k0, lane);
    v16h a1 = load_frag_k16(Az, lda, row0 + 16, k0, lane);
    v16h b0 = load_frag_k16(Bz, ldb, col0,      k0, lane);
    v16h b1 = load_frag_k16(Bz, ldb, col0 + 16, k0, lane);
    if (k0 + 32 < K) {  // warm L0 for the next K-step (global_prefetch_b8)
      __builtin_prefetch((const void*)(Az + (size_t)(row0 + rA) * lda + (k0 + 32 + kgo)), 0, 1);
      __builtin_prefetch((const void*)(Bz + (size_t)(col0 + rA) * ldb + (k0 + 32 + kgo)), 0, 1);
    }
    acc[0][0].v = __builtin_amdgcn_wmma_f32_16x16x32_f16(false, a0, false, b0, (short)0, acc[0][0].v, false, false);
    acc[0][1].v = __builtin_amdgcn_wmma_f32_16x16x32_f16(false, a0, false, b1, (short)0, acc[0][1].v, false, false);
    acc[1][0].v = __builtin_amdgcn_wmma_f32_16x16x32_f16(false, a1, false, b0, (short)0, acc[1][0].v, false, false);
    acc[1][1].v = __builtin_amdgcn_wmma_f32_16x16x32_f16(false, a1, false, b1, (short)0, acc[1][1].v, false, false);
  }

  // C/D layout: elem r, lanes 0-15 -> M=r, N=lane ; lanes16-31 -> M=8+r, N=lane-16
  int rbase = (lane >> 4) * 8;
  int cl    = lane & 15;
#pragma unroll
  for (int i = 0; i < 2; ++i)
#pragma unroll
    for (int j = 0; j < 2; ++j)
#pragma unroll
      for (int r = 0; r < 8; ++r) {
        int row = row0 + i * 16 + rbase + r;
        int col = col0 + j * 16 + cl;
        float v = acc[i][j].f[r];
        if (bias)  v += bias[col];
        long long idx = cOff + (long long)row * ldc + col;
        if (resid) v += resid[idx];
        if (doRelu) v = v > 0.0f ? v : 0.0f;
        if (Cf) Cf[idx] = v;
        if (Ch) {
          long long hidx = storeCT ? (cOff + (long long)col * ldc + row) : idx;
          Ch[hidx] = (_Float16)v;
        }
      }
}

// ---------------- fp32 [K,N] -> f16 transposed [N,K], LDS-tiled ----------------
__global__ __launch_bounds__(256)
void k_transpose(const float* __restrict__ W, _Float16* __restrict__ Wt, int K, int N)
{
  __shared__ float tile[32][33];
  int kt = blockIdx.y * 32, nt = blockIdx.x * 32;
  int tx = threadIdx.x & 31, ty = threadIdx.x >> 5;   // 8 rows per pass
  for (int r = ty; r < 32; r += 8) {
    int k = kt + r, n = nt + tx;
    tile[r][tx] = (k < K && n < N) ? W[(size_t)k * N + n] : 0.0f;
  }
  __syncthreads();
  for (int r = ty; r < 32; r += 8) {
    int n = nt + r, k = kt + tx;
    if (n < N && k < K) Wt[(size_t)n * K + k] = (_Float16)tile[tx][r];
  }
}

// ---------------- embedding + sinusoidal positional encoding ----------------
__global__ __launch_bounds__(256)
void k_embed(const int* __restrict__ tok, const float* __restrict__ emb,
             float* __restrict__ xf, _Float16* __restrict__ xh)
{
  int bs = blockIdx.x;            // [0, B*S)
  int t  = tok[bs];
  int s  = bs & (S_ - 1);
  const float* e = emb + (size_t)t * D_;
  for (int d = threadIdx.x; d < D_; d += 256) {
    float ang = (float)s * __expf(-9.2103403719761836f * (float)(d & ~1) / (float)D_);
    float pe  = (d & 1) ? __cosf(ang) : __sinf(ang);
    float v   = e[d] + pe;
    xf[(size_t)bs * D_ + d] = v;
    xh[(size_t)bs * D_ + d] = (_Float16)v;
  }
}

// ---------------- wave-per-row masked softmax over S=128 keys ----------------
__global__ __launch_bounds__(256)
void k_softmax(float* __restrict__ sc, _Float16* __restrict__ attnH,
               const int* __restrict__ tok, int Hn)
{
  int wave = threadIdx.x >> 5, lane = threadIdx.x & 31;
  int row  = blockIdx.x * 8 + wave;          // [0, B*Hn*S)
  int b    = row / (Hn * S_);
  float* p = sc + (long long)row * S_;
  float v[4];
#pragma unroll
  for (int t = 0; t < 4; ++t) {
    int j = lane + t * 32;
    bool pad = (tok[b * S_ + j] == 0);
    v[t] = pad ? -1e9f : p[j] * 0.125f;      // 1/sqrt(DK), DK=64
  }
  float mx = fmaxf(fmaxf(v[0], v[1]), fmaxf(v[2], v[3]));
#pragma unroll
  for (int o = 16; o > 0; o >>= 1) mx = fmaxf(mx, __shfl_xor(mx, o, 32));
  float sum = 0.0f;
#pragma unroll
  for (int t = 0; t < 4; ++t) { v[t] = __expf(v[t] - mx); sum += v[t]; }
#pragma unroll
  for (int o = 16; o > 0; o >>= 1) sum += __shfl_xor(sum, o, 32);
  float inv = 1.0f / sum;
#pragma unroll
  for (int t = 0; t < 4; ++t) {
    int j = lane + t * 32;
    float a = v[t] * inv;
    p[j] = a;
    if (attnH) attnH[(long long)row * S_ + j] = (_Float16)a;
  }
}

// ---------------- column sum over heads & queries: dst[b,s] ----------------
__global__ void k_colsum(const float* __restrict__ attn, float* __restrict__ dst, int Hn)
{
  int b = blockIdx.x, s = threadIdx.x;       // 128 threads
  float acc = 0.0f;
  for (int h = 0; h < Hn; ++h) {
    const float* p = attn + (((long long)b * Hn + h) * S_) * S_ + s;
    for (int q = 0; q < S_; ++q) acc += p[(long long)q * S_];
  }
  dst[b * S_ + s] = acc;
}

// ---------------- creat_mask: drop lowest-attention 20% of prefix ----------------
__global__ __launch_bounds__(128)
void k_creat_mask(const float* __restrict__ col, const int* __restrict__ tokens,
                  int* __restrict__ mtok, float* __restrict__ outTok,
                  float* __restrict__ outMtok)
{
  __shared__ float vals[S_];
  __shared__ int sj0;
  int b = blockIdx.x, j = threadIdx.x;
  int t = tokens[b * S_ + j];
  if (j == 0) sj0 = S_;
  __syncthreads();
  if (t == 0) atomicMin(&sj0, j);
  __syncthreads();
  int j0 = sj0;
  int kk = (j0 < S_) ? (int)ceilf(0.2f * (float)j0) : 0;
  float v = (j < j0) ? col[b * S_ + j] : INFINITY;
  vals[j] = v;
  __syncthreads();
  int rank = 0;
  for (int l = 0; l < S_; ++l) {
    float w = vals[l];
    rank += (w < v) || (w == v && l < j);    // stable double-argsort rank
  }
  int keep = (rank >= kk) ? 1 : 0;
  int mt = t * keep;
  mtok[b * S_ + j]    = mt;
  outTok[b * S_ + j]  = (float)t;
  outMtok[b * S_ + j] = (float)mt;
}

// ---------------- LayerNorm (identity affine) over D ----------------
__global__ __launch_bounds__(256)
void k_layernorm(const float* __restrict__ in, float* __restrict__ outF,
                 _Float16* __restrict__ outH, int Dd)
{
  __shared__ float red[256];
  long long row = blockIdx.x;
  const float* p = in + row * Dd;
  float s = 0.0f;
  for (int i = threadIdx.x; i < Dd; i += 256) s += p[i];
  red[threadIdx.x] = s; __syncthreads();
  for (int o = 128; o > 0; o >>= 1) { if (threadIdx.x < o) red[threadIdx.x] += red[threadIdx.x + o]; __syncthreads(); }
  float mean = red[0] / (float)Dd;
  __syncthreads();
  float vs = 0.0f;
  for (int i = threadIdx.x; i < Dd; i += 256) { float d = p[i] - mean; vs += d * d; }
  red[threadIdx.x] = vs; __syncthreads();
  for (int o = 128; o > 0; o >>= 1) { if (threadIdx.x < o) red[threadIdx.x] += red[threadIdx.x + o]; __syncthreads(); }
  float rstd = rsqrtf(red[0] / (float)Dd + 1e-5f);
  for (int i = threadIdx.x; i < Dd; i += 256) {
    float y = (p[i] - mean) * rstd;
    outF[row * Dd + i] = y;
    outH[row * Dd + i] = (_Float16)y;
  }
}

// ===========================================================================
extern "C" void kernel_launch(void* const* d_in, const int* in_sizes, int n_in,
                              void* d_out, int out_size, void* d_ws, size_t ws_size,
                              hipStream_t stream)
{
  (void)in_sizes; (void)n_in; (void)out_size; (void)ws_size;
  const int*   tokens = (const int*)  d_in[0];
  const float* emb    = (const float*)d_in[1];
  const float* aWQ    = (const float*)d_in[2];
  const float* aWK    = (const float*)d_in[3];
  const float* tWQ    = (const float*)d_in[8];
  const float* tWK    = (const float*)d_in[9];
  const float* tWV    = (const float*)d_in[10];
  const float* tFC    = (const float*)d_in[11];
  const float* tF1    = (const float*)d_in[12];
  const float* tF2    = (const float*)d_in[13];
  const float* pW1    = (const float*)d_in[14];
  const float* pb1    = (const float*)d_in[15];
  const float* pW2    = (const float*)d_in[16];
  const float* pb2    = (const float*)d_in[17];
  const float* pW3    = (const float*)d_in[18];
  const float* pb3    = (const float*)d_in[19];

  float* out       = (float*)d_out;
  float* outTok    = out;                 // B*S
  float* out1      = out + 8192;          // B*128
  float* outMtokF  = out + 16384;         // B*S
  float* outM      = out + 24576;         // B*128
  float* outScores = out + 32768;         // L*B*S

  // ---- workspace carving ----
  char* wsb = (char*)d_ws;
  size_t off = 0;
  auto allocB = [&](size_t bytes) -> char* { char* p = wsb + off; off = (off + bytes + 255) & ~(size_t)255; return p; };
  auto allocH = [&](size_t n) -> _Float16* { return (_Float16*)allocB(n * sizeof(_Float16)); };
  auto allocF = [&](size_t n) -> float*    { return (float*)allocB(n * sizeof(float)); };

  const size_t MT = (size_t)B_ * S_;      // 8192 rows
  _Float16* aWQt = allocH(128 * 512);
  _Float16* aWKt = allocH(128 * 512);
  _Float16* tWQt = allocH((size_t)L_ * 512 * 512);
  _Float16* tWKt = allocH((size_t)L_ * 512 * 512);
  _Float16* tWVt = allocH((size_t)L_ * 512 * 512);
  _Float16* tFCt = allocH((size_t)L_ * 512 * 512);
  _Float16* tF1t = allocH((size_t)L_ * 2048 * 512);
  _Float16* tF2t = allocH((size_t)L_ * 512 * 2048);
  _Float16* pW1t = allocH((size_t)1024 * 65536);
  _Float16* pW2t = allocH((size_t)512 * 1024);
  _Float16* pW3t = allocH((size_t)128 * 512);

  float*    xf      = allocF(MT * D_);
  _Float16* xh      = allocH(MT * D_);
  _Float16* qh      = allocH(MT * D_);
  _Float16* kh      = allocH(MT * D_);
  _Float16* vth     = allocH(MT * D_);    // V stored transposed: [B][H*DK][S]
  float*    scoresf = allocF((size_t)B_ * H_ * S_ * S_);
  _Float16* attnh   = allocH((size_t)B_ * H_ * S_ * S_);
  _Float16* ctxh    = allocH(MT * D_);
  float*    tmpf    = allocF(MT * D_);
  _Float16* ffnh    = allocH(MT * DFF_);
  float*    colf    = allocF(MT);
  int*      mtok    = (int*)allocB(MT * sizeof(int));
  _Float16* gh1     = allocH((size_t)B_ * 1024);
  _Float16* gh2     = allocH((size_t)B_ * 512);

  auto transpose = [&](const float* W, _Float16* Wt, int K, int N) {
    dim3 g((N + 31) / 32, (K + 31) / 32, 1);
    k_transpose<<<g, dim3(256), 0, stream>>>(W, Wt, K, N);
  };
  auto gemm = [&](const _Float16* A, const _Float16* Bt, const float* bias, const float* resid,
                  float* Cf, _Float16* Ch, int M, int N, int K, int lda, int ldb, int ldc,
                  int Z, int Z2, long long sA1, long long sA2, long long sB1, long long sB2,
                  long long sC1, long long sC2, int storeCT, int relu) {
    dim3 grid((N + 127) / 128, (M + 63) / 64, Z);
    k_gemm_wmma<<<grid, dim3(256), 0, stream>>>(A, Bt, bias, resid, Cf, Ch, M, N, K,
                                                lda, ldb, ldc, Z2, sA1, sA2, sB1, sB2,
                                                sC1, sC2, storeCT, relu);
  };

  // ---- one-time weight convert+transpose (f32 [K,N] -> f16 [N,K]) ----
  transpose(aWQ, aWQt, 512, 128);
  transpose(aWK, aWKt, 512, 128);
  for (int l = 0; l < L_; ++l) {
    transpose(tWQ + (size_t)l * 512 * 512,  tWQt + (size_t)l * 512 * 512,  512, 512);
    transpose(tWK + (size_t)l * 512 * 512,  tWKt + (size_t)l * 512 * 512,  512, 512);
    transpose(tWV + (size_t)l * 512 * 512,  tWVt + (size_t)l * 512 * 512,  512, 512);
    transpose(tFC + (size_t)l * 512 * 512,  tFCt + (size_t)l * 512 * 512,  512, 512);
    transpose(tF1 + (size_t)l * 512 * 2048, tF1t + (size_t)l * 2048 * 512, 512, 2048);
    transpose(tF2 + (size_t)l * 2048 * 512, tF2t + (size_t)l * 512 * 2048, 2048, 512);
  }
  transpose(pW1, pW1t, 65536, 1024);
  transpose(pW2, pW2t, 1024, 512);
  transpose(pW3, pW3t, 512, 128);

  // ---- embed original tokens ----
  k_embed<<<dim3(B_ * S_), dim3(256), 0, stream>>>(tokens, emb, xf, xh);

  // ---- mask-generating 2-head attention (only Q,K,scores needed) ----
  gemm(xh, aWQt, nullptr, nullptr, nullptr, qh, 8192, 128, 512, 512, 512, 128,
       1, 1, 0, 0, 0, 0, 0, 0, 0, 0);
  gemm(xh, aWKt, nullptr, nullptr, nullptr, kh, 8192, 128, 512, 512, 512, 128,
       1, 1, 0, 0, 0, 0, 0, 0, 0, 0);
  gemm(qh, kh, nullptr, nullptr, scoresf, nullptr, 128, 128, 64, 128, 128, 128,
       B_ * AH_, AH_, (long long)S_ * 128, 64, (long long)S_ * 128, 64,
       (long long)AH_ * S_ * S_, (long long)S_ * S_, 0, 0);
  k_softmax<<<dim3(B_ * AH_ * S_ / 8), dim3(256), 0, stream>>>(scoresf, (_Float16*)nullptr, tokens, AH_);
  k_colsum<<<dim3(B_), dim3(128), 0, stream>>>(scoresf, colf, AH_);
  k_creat_mask<<<dim3(B_), dim3(128), 0, stream>>>(colf, tokens, mtok, outTok, outMtokF);

  // ---- 6-layer encoder pass + pre-head ----
  auto encode_pass = [&](const int* tokPtr, float* scoresOut, float* headOut) {
    for (int l = 0; l < L_; ++l) {
      const _Float16* WQt = tWQt + (size_t)l * 512 * 512;
      const _Float16* WKt = tWKt + (size_t)l * 512 * 512;
      const _Float16* WVt = tWVt + (size_t)l * 512 * 512;
      const _Float16* FCt = tFCt + (size_t)l * 512 * 512;
      const _Float16* F1t = tF1t + (size_t)l * 2048 * 512;
      const _Float16* F2t = tF2t + (size_t)l * 512 * 2048;
      // Q, K projections (natural layout)
      gemm(xh, WQt, nullptr, nullptr, nullptr, qh, 8192, 512, 512, 512, 512, 512,
           1, 1, 0, 0, 0, 0, 0, 0, 0, 0);
      gemm(xh, WKt, nullptr, nullptr, nullptr, kh, 8192, 512, 512, 512, 512, 512,
           1, 1, 0, 0, 0, 0, 0, 0, 0, 0);
      // V projection, stored transposed per batch row-block: vth[b, h*64+d, s]
      gemm(xh, WVt, nullptr, nullptr, nullptr, vth, 128, 512, 512, 512, 512, 128,
           B_, 1, (long long)S_ * 512, 0, 0, 0, (long long)S_ * 512, 0, 1, 0);
      // scores = Q * K^T, batched over (b,h)
      gemm(qh, kh, nullptr, nullptr, scoresf, nullptr, 128, 128, 64, 512, 512, 128,
           B_ * H_, H_, (long long)S_ * 512, 64, (long long)S_ * 512, 64,
           (long long)H_ * S_ * S_, (long long)S_ * S_, 0, 0);
      k_softmax<<<dim3(B_ * H_ * S_ / 8), dim3(256), 0, stream>>>(scoresf, attnh, tokPtr, H_);
      if (scoresOut)
        k_colsum<<<dim3(B_), dim3(128), 0, stream>>>(scoresf, scoresOut + (size_t)l * B_ * S_, H_);
      // ctx = attn * V   (B operand = vth, contiguous fragments), batched over (b,h)
      gemm(attnh, vth, nullptr, nullptr, nullptr, ctxh, 128, 64, 128, 128, 128, 512,
           B_ * H_, H_, (long long)H_ * S_ * S_, (long long)S_ * S_,
           (long long)S_ * 512, (long long)DK_ * S_, (long long)S_ * 512, 64, 0, 0);
      // out = ctx @ WO + x ; LN
      gemm(ctxh, FCt, nullptr, xf, tmpf, nullptr, 8192, 512, 512, 512, 512, 512,
           1, 1, 0, 0, 0, 0, 0, 0, 0, 0);
      k_layernorm<<<dim3(8192), dim3(256), 0, stream>>>(tmpf, xf, xh, 512);
      // FFN
      gemm(xh, F1t, nullptr, nullptr, nullptr, ffnh, 8192, 2048, 512, 512, 512, 2048,
           1, 1, 0, 0, 0, 0, 0, 0, 0, 1);
      gemm(ffnh, F2t, nullptr, xf, tmpf, nullptr, 8192, 512, 2048, 2048, 2048, 512,
           1, 1, 0, 0, 0, 0, 0, 0, 0, 0);
      k_layernorm<<<dim3(8192), dim3(256), 0, stream>>>(tmpf, xf, xh, 512);
    }
    // pre_head MLP on [64, 65536]
    gemm(xh, pW1t, pb1, nullptr, nullptr, gh1, 64, 1024, 65536, 65536, 65536, 1024,
         1, 1, 0, 0, 0, 0, 0, 0, 0, 1);
    gemm(gh1, pW2t, pb2, nullptr, nullptr, gh2, 64, 512, 1024, 1024, 1024, 512,
         1, 1, 0, 0, 0, 0, 0, 0, 0, 1);
    gemm(gh2, pW3t, pb3, nullptr, headOut, nullptr, 64, 128, 512, 512, 512, 128,
         1, 1, 0, 0, 0, 0, 0, 0, 0, 0);
  };

  // pass 1: original tokens (xf/xh already hold embed(tokens))
  encode_pass(tokens, outScores, out1);

  // pass 2: masked tokens
  k_embed<<<dim3(B_ * S_), dim3(256), 0, stream>>>(mtok, emb, xf, xh);
  encode_pass(mtok, nullptr, outM);
}
